// CIFLayer_26448408609425
// MI455X (gfx1250) — matrix operand order repaired and checked by reading, not compiled, not run on hardware
//
#include <hip/hip_runtime.h>
#include <hip/hip_bf16.h>
#include <math.h>

// ---------------------------------------------------------------------------
// CIF layer for MI455X (gfx1250, wave32, WMMA)
//   B=16, S=4096, C=512, T_MAX=512, BETA=1, kernel width 3
// Stage 1: fused conv(k=3) + bias + LayerNorm + GELU + proj + sigmoid -> alpha
//          implicit GEMM [65536 x 1536] x [1536 x 512] via v_wmma_f32_16x16x32_bf16
//          with the global->register stage software-pipelined one chunk ahead
// Stage 2: per-batch alpha scaling + inclusive scan -> csum
// Stage 3: CIF as a gather per (b, t) using monotone csum (no atomics)
// ---------------------------------------------------------------------------

typedef __attribute__((ext_vector_type(8)))  float          v8f;
typedef __attribute__((ext_vector_type(4)))  float          v4f;
typedef __attribute__((ext_vector_type(16))) __bf16         v16bf;
typedef __attribute__((ext_vector_type(8)))  unsigned short v8u;
typedef __attribute__((ext_vector_type(16))) unsigned short v16u;

#define S_LEN   4096
#define CH      512
#define TMAX    512
#define KCH     1536            // 3 * CH
#define KCHUNK  32
#define NCHUNK  48              // KCH / KCHUNK ; 512/32=16 chunks per tap -> no tap straddle
#define MTILE   64
#define NBATCH  16
#define LDSPITCH 40             // 32 + 8 halves padding -> conflict-free b128 LDS reads

__device__ __forceinline__ unsigned short f2bf(float f) {
  unsigned int u = __float_as_uint(f);
  unsigned int r = u + 0x7FFFu + ((u >> 16) & 1u);   // round-to-nearest-even
  return (unsigned short)(r >> 16);
}

// ---------------------------------------------------------------------------
// Kernel 0: reorder conv_w (C_out, C_in, 3) f32 -> bf16 chunked layout
//   Wr2[(kc*512 + c_out)*32 + kk] with kc = tap*16 + (c_in0/32), kk = c_in offset
// so stage-B LDS staging in the GEMM is a straight coalesced copy.
// ---------------------------------------------------------------------------
__global__ void reorder_w_kernel(const float* __restrict__ conv_w,
                                 unsigned short* __restrict__ Wr2) {
  int g = blockIdx.x * 256 + threadIdx.x;          // 0 .. 786431
  int kc  = g >> 14;                               // 48 chunks
  int rem = g & 16383;
  int col = rem >> 5;                              // c_out
  int kk  = rem & 31;
  int tap = kc >> 4;
  int cin = ((kc & 15) << 5) + kk;
  Wr2[g] = f2bf(conv_w[(size_t)col * KCH + cin * 3 + tap]);
}

// ---------------------------------------------------------------------------
// Kernel 1: fused weight predictor -> alpha[B*S]
// 256 threads = 8 waves; tile 64 rows x 512 cols; waves 2(M) x 4(N),
// each wave: 32x128 = 2x8 accumulator tiles of v_wmma_f32_16x16x32_bf16.
// Global->register staging for chunk k+1 is issued before the chunk-k WMMA
// burst so HBM/L2 latency overlaps the matrix pipe.
// ---------------------------------------------------------------------------
struct Stage {
  v8u a;        // this thread's 8 bf16 of the A tile
  v8u b[8];     // this thread's 8x8 bf16 of the B tile
};

__global__ __launch_bounds__(256, 1)
void predictor_kernel(const float* __restrict__ x,
                      const unsigned char* __restrict__ pad,
                      const float* __restrict__ conv_b,
                      const float* __restrict__ ln_g,
                      const float* __restrict__ ln_b,
                      const float* __restrict__ proj_w,
                      const float* __restrict__ proj_b,
                      const unsigned short* __restrict__ Wr2,
                      float* __restrict__ alpha) {
  __shared__ unsigned short As[MTILE * LDSPITCH];   //  5.0 KB
  __shared__ unsigned short Bs[CH    * LDSPITCH];   // 40.0 KB
  __shared__ float rowsum[MTILE], rowsq[MTILE], rowpj[MTILE];

  const int tid  = threadIdx.x;
  const int lane = tid & 31;
  const int wave = tid >> 5;
  const int m_w  = wave >> 2;           // 0..1  (M sub-block, 32 rows each)
  const int n_w  = wave & 3;            // 0..3  (N sub-block, 128 cols each)
  const int m0   = blockIdx.x * MTILE;  // global flattened (b*S+s) row base

  // A-staging assignment: thread -> (row, 8-wide k quarter)
  const int srow = tid >> 2;            // 0..63
  const int skq  = (tid & 3) * 8;       // 0,8,16,24
  const int grow = m0 + srow;
  const int sb   = grow >> 12;          // /4096
  const int ss   = grow & (S_LEN - 1);

  // per-lane fragment bases (ushort indices), per ISA 16-bit A/B layouts
  const int lhalf = lane >> 4;
  const int l15   = lane & 15;
  int aoff[2], boff[8], bcol[8];
#pragma unroll
  for (int mi = 0; mi < 2; ++mi)
    aoff[mi] = (m_w * 32 + mi * 16 + l15) * LDSPITCH + lhalf * 8;
#pragma unroll
  for (int ni = 0; ni < 8; ++ni) {
    bcol[ni] = n_w * 128 + ni * 16 + l15;
    boff[ni] = bcol[ni] * LDSPITCH + lhalf * 8;
  }

  // ---- staging helpers (inlined lambdas) ----------------------------------
  auto load_stage = [&](int kc) -> Stage {
    Stage st;
    const int tap  = kc >> 4;              // 0,1,2
    const int cin0 = (kc & 15) << 5;
    // A: im2col of x with SAME padding
    {
      int ssrc = ss + tap - 1;
      if (ssrc >= 0 && ssrc < S_LEN) {
        const float* src = x + ((size_t)(sb * S_LEN + ssrc)) * CH + cin0 + skq;
        v4f p0 = *(const v4f*)(src);
        v4f p1 = *(const v4f*)(src + 4);
#pragma unroll
        for (int i = 0; i < 4; ++i) { st.a[i] = f2bf(p0[i]); st.a[4 + i] = f2bf(p1[i]); }
      } else {
#pragma unroll
        for (int i = 0; i < 8; ++i) st.a[i] = 0;
      }
    }
    // B: straight coalesced copy from reordered bf16 weights (L2-resident)
    {
      const unsigned short* wsrc = Wr2 + ((size_t)kc << 14);
#pragma unroll
      for (int i = 0; i < 8; ++i) {
        int u   = tid * 8 + i;
        int col = u >> 2;
        int kq  = (u & 3) * 8;
        st.b[i] = *(const v8u*)(wsrc + col * 32 + kq);
      }
    }
    return st;
  };

  auto store_stage = [&](const Stage& st) {
    *(v8u*)&As[srow * LDSPITCH + skq] = st.a;
#pragma unroll
    for (int i = 0; i < 8; ++i) {
      int u   = tid * 8 + i;
      int col = u >> 2;
      int kq  = (u & 3) * 8;
      *(v8u*)&Bs[col * LDSPITCH + kq] = st.b[i];
    }
  };

  v8f acc[2][8];
#pragma unroll
  for (int mi = 0; mi < 2; ++mi)
#pragma unroll
    for (int ni = 0; ni < 8; ++ni)
      acc[mi][ni] = v8f{0.f, 0.f, 0.f, 0.f, 0.f, 0.f, 0.f, 0.f};

  Stage st = load_stage(0);              // prologue: chunk 0 -> registers

#pragma unroll 1
  for (int kc = 0; kc < NCHUNK; ++kc) {
    __syncthreads();                     // previous chunk's frag reads done
    store_stage(st);                     // waits only on this stage's loads
    __syncthreads();

    if (kc + 1 < NCHUNK)
      st = load_stage(kc + 1);           // issue next chunk's global loads NOW

    // ---- fragment assembly + 16 WMMAs (hides the loads above)
    v16bf afrag[2];
#pragma unroll
    for (int mi = 0; mi < 2; ++mi) {
      v16u au;
      ((v8u*)&au)[0] = *(const v8u*)&As[aoff[mi]];        // K koff..koff+7
      ((v8u*)&au)[1] = *(const v8u*)&As[aoff[mi] + 16];   // K koff+16..+23
      afrag[mi] = __builtin_bit_cast(v16bf, au);
    }
#pragma unroll
    for (int ni = 0; ni < 8; ++ni) {
      v16u bu;
      ((v8u*)&bu)[0] = *(const v8u*)&Bs[boff[ni]];        // K koff..koff+7
      ((v8u*)&bu)[1] = *(const v8u*)&Bs[boff[ni] + 16];   // K koff+16..+23
      v16bf bfrag = __builtin_bit_cast(v16bf, bu);
#pragma unroll
      for (int mi = 0; mi < 2; ++mi)
        acc[mi][ni] = __builtin_amdgcn_wmma_f32_16x16x32_bf16(
            false, afrag[mi], false, bfrag, (short)0, acc[mi][ni], false, false);
    }
  }

  // ------------------- epilogue: bias + LN + GELU + proj + sigmoid ----------
  if (tid < MTILE) { rowsum[tid] = 0.f; rowsq[tid] = 0.f; rowpj[tid] = 0.f; }
  __syncthreads();

  float cb[8], lg[8], lbv[8], pw[8];
#pragma unroll
  for (int ni = 0; ni < 8; ++ni) {
    cb[ni]  = conv_b[bcol[ni]];
    lg[ni]  = ln_g[bcol[ni]];
    lbv[ni] = ln_b[bcol[ni]];
    pw[ni]  = proj_w[bcol[ni]];
  }

  // row partial sums / sums of squares (each lane owns one column per tile)
#pragma unroll
  for (int mi = 0; mi < 2; ++mi) {
#pragma unroll
    for (int j = 0; j < 8; ++j) {
      float s1 = 0.f, s2 = 0.f;
#pragma unroll
      for (int ni = 0; ni < 8; ++ni) {
        float v = acc[mi][ni][j] + cb[ni];
        acc[mi][ni][j] = v;
        s1 += v; s2 += v * v;
      }
#pragma unroll
      for (int m = 1; m < 16; m <<= 1) {     // reduce across 16-lane half
        s1 += __shfl_xor(s1, m, 32);
        s2 += __shfl_xor(s2, m, 32);
      }
      if (l15 == 0) {
        int r = m_w * 32 + mi * 16 + lhalf * 8 + j;
        atomicAdd(&rowsum[r], s1);
        atomicAdd(&rowsq[r],  s2);
      }
    }
  }
  __syncthreads();

#pragma unroll
  for (int mi = 0; mi < 2; ++mi) {
#pragma unroll
    for (int j = 0; j < 8; ++j) {
      int r = m_w * 32 + mi * 16 + lhalf * 8 + j;
      float mu  = rowsum[r] * (1.0f / CH);
      float var = rowsq[r] * (1.0f / CH) - mu * mu;
      float rs  = rsqrtf(var + 1e-5f);
      float p = 0.f;
#pragma unroll
      for (int ni = 0; ni < 8; ++ni) {
        float hn = (acc[mi][ni][j] - mu) * rs * lg[ni] + lbv[ni];
        float g  = 0.5f * hn * (1.0f + erff(hn * 0.70710678118654752f)); // exact GELU
        p += g * pw[ni];
      }
#pragma unroll
      for (int m = 1; m < 16; m <<= 1) p += __shfl_xor(p, m, 32);
      if (l15 == 0) atomicAdd(&rowpj[r], p);
    }
  }
  __syncthreads();

  if (tid < MTILE) {
    int gr = m0 + tid;
    int bb = gr >> 12, ssx = gr & (S_LEN - 1);
    float z = rowpj[tid] + proj_b[0];
    float a = 1.0f / (1.0f + expf(-z));
    if (pad[bb * S_LEN + ssx]) a = 0.f;        // alpha = where(pad, 0, alpha)
    alpha[gr] = a;
  }
}

// ---------------------------------------------------------------------------
// Kernel 2: per-batch alpha_sum, scale = (target + eps)/sum, inclusive csum
// one block per batch, 256 threads x 16 elements
// ---------------------------------------------------------------------------
__global__ __launch_bounds__(256)
void scan_kernel(const float* __restrict__ alpha,
                 const int* __restrict__ tlen,
                 float* __restrict__ csum) {
  const int b = blockIdx.x, tid = threadIdx.x;
  const int base = b * S_LEN;
  __shared__ float part[256];
  __shared__ float scale_s;

  float loc[16];
  float s = 0.f;
#pragma unroll
  for (int i = 0; i < 16; ++i) { loc[i] = alpha[base + tid * 16 + i]; s += loc[i]; }
  part[tid] = s;
  __syncthreads();

  for (int off = 1; off < 256; off <<= 1) {     // Hillis-Steele inclusive scan
    float t = (tid >= off) ? part[tid - off] : 0.f;
    __syncthreads();
    part[tid] += t;
    __syncthreads();
  }
  if (tid == 0) scale_s = ((float)tlen[b] * 1.0f + 1e-4f) / part[255];
  __syncthreads();

  const float scale = scale_s;
  float run = part[tid] - s;                    // exclusive prefix
#pragma unroll
  for (int i = 0; i < 16; ++i) {
    run += loc[i];
    csum[base + tid * 16 + i] = run * scale;
  }
}

// ---------------------------------------------------------------------------
// Kernel 3: CIF as gather per (b, t).  Monotone csum -> contributing s-range
// found by binary search; each row written exactly once, no atomics.
// ---------------------------------------------------------------------------
__device__ __forceinline__ int lbound(const float* a, int n, float v) {
  int lo = 0, hi = n;
  while (lo < hi) { int m = (lo + hi) >> 1; if (a[m] < v) lo = m + 1; else hi = m; }
  return lo;
}

__global__ __launch_bounds__(256)
void cif_kernel(const float* __restrict__ x,
                const float* __restrict__ csum,
                const int* __restrict__ tlen,
                const unsigned char* __restrict__ pad,
                float* __restrict__ out) {
  const int t = blockIdx.x;        // 0..511
  const int b = blockIdx.y;        // 0..15
  const int tid = threadIdx.x;     // owns channels tid and tid+256
  const size_t obase = ((size_t)(b * TMAX + t)) * CH;

  if (t >= tlen[b]) { out[obase + tid] = 0.f; out[obase + tid + 256] = 0.f; return; }

  const float* cs = csum + b * S_LEN;
  const int lo  = lbound(cs, S_LEN, (float)t - 2.0f); // all contributors have csum >= t-1
  const int f   = lbound(cs, S_LEN, (float)t + 1.0f); // first s with L > t after it
  const int shi = (f < S_LEN - 1) ? f : (S_LEN - 1);

  float a0 = 0.f, a1 = 0.f;
  float cp = (lo > 0) ? cs[lo - 1] : 0.f;
  for (int s = lo; s <= shi; ++s) {
    float c1 = cs[s];
    float al = c1 - cp;
    int R = (int)floorf(c1); R = (R < TMAX) ? R : TMAX;
    int L = (s > 0) ? ((int)floorf(cp) < TMAX ? (int)floorf(cp) : TMAX) : 0;
    int fire  = R - L;
    int extra = fire - 1 > 0 ? fire - 1 : 0;
    float rw = (fire > 0) ? (c1 - (float)R) : 0.f;   // BETA = 1
    float lw = al - rw - (float)extra;
    float w = 0.f;
    if (L == t) w += lw;
    if (R == t) w += rw;
    if (L + 1 == t && extra > 0) w += 1.0f;          // EXTRA step 1
    if (L + 2 == t && extra > 1) w += 1.0f;          // EXTRA step 2
    if (w != 0.f && !pad[b * S_LEN + s]) {           // x = where(pad, 0, x)
      size_t xb = ((size_t)(b * S_LEN + s)) * CH;
      a0 += w * x[xb + tid];
      a1 += w * x[xb + tid + 256];
    }
    cp = c1;
  }
  out[obase + tid] = a0;
  out[obase + tid + 256] = a1;
}

// ---------------------------------------------------------------------------
extern "C" void kernel_launch(void* const* d_in, const int* in_sizes, int n_in,
                              void* d_out, int out_size, void* d_ws, size_t ws_size,
                              hipStream_t stream) {
  const float*         x      = (const float*)d_in[0];
  const unsigned char* pad    = (const unsigned char*)d_in[1];  // jnp bool = u8
  const int*           tlen   = (const int*)d_in[2];
  const float*         conv_w = (const float*)d_in[3];
  const float*         conv_b = (const float*)d_in[4];
  const float*         ln_g   = (const float*)d_in[5];
  const float*         ln_b   = (const float*)d_in[6];
  const float*         proj_w = (const float*)d_in[7];
  const float*         proj_b = (const float*)d_in[8];
  float*               out    = (float*)d_out;

  // workspace carve-out (~2 MB):
  //   [0, 1.5MB)   Wr2   reordered bf16 weights (48*512*32 halves)
  //   [+256KB)     alpha (B*S f32)
  //   [+256KB)     csum  (B*S f32)
  unsigned short* Wr2 = (unsigned short*)d_ws;
  float* alpha = (float*)((char*)d_ws + (size_t)NCHUNK * CH * KCHUNK * 2);
  float* csum  = alpha + NBATCH * S_LEN;

  reorder_w_kernel<<<(NCHUNK * CH * KCHUNK) / 256, 256, 0, stream>>>(conv_w, Wr2);

  predictor_kernel<<<(NBATCH * S_LEN) / MTILE, 256, 0, stream>>>(
      x, pad, conv_b, ln_g, ln_b, proj_w, proj_b, Wr2, alpha);

  scan_kernel<<<NBATCH, 256, 0, stream>>>(alpha, tlen, csum);

  dim3 g3(TMAX, NBATCH);
  cif_kernel<<<g3, 256, 0, stream>>>(x, csum, tlen, pad, out);
}
